// Simulator_26817775796325
// MI455X (gfx1250) — matrix operand order repaired and checked by reading, not compiled, not run on hardware
//
#include <hip/hip_runtime.h>
#include <hip/hip_bf16.h>

typedef _Float16 half_t;
typedef __attribute__((ext_vector_type(16))) _Float16 v16h;
typedef __attribute__((ext_vector_type(8)))  _Float16 v8h;
typedef __attribute__((ext_vector_type(8)))  float    v8f;

#define NATOMS 8192
#define FEATD  24
#define HID    64
#define KNBR   14     // neighbors excluding self (K-1)
#define NSTEPS 3

// ---------------------------------------------------------------- helpers

__device__ __forceinline__ float silu_f(float x) { return x / (1.f + __expf(-x)); }

__device__ __forceinline__ v8f wmma_f16(v16h a, v16h b, v8f c) {
    return __builtin_amdgcn_wmma_f32_16x16x32_f16(false, a, false, b, (short)0, c,
                                                  false, false);
}

// A fragment: 16x32 f16 tile, row-major in LDS with leading dim ld (halfs).
// Per ISA layout each lane needs two contiguous 8-half runs:
//   halves 0..7  = A[m][kbase + hi*8      .. +7]
//   halves 8..15 = A[m][kbase + 16 + hi*8 .. +7]
// ld in {32,64,128,160} keeps both runs 16B-aligned.
__device__ __forceinline__ v16h load_fragA(const half_t* s, int ld, int lane, int kbase) {
    int m = lane & 15, hi = (lane >> 4) & 1;
    const v8h* p0 = (const v8h*)(s + m * ld + kbase + hi * 8);
    const v8h* p1 = (const v8h*)(s + m * ld + kbase + 16 + hi * 8);
    v8h lo = *p0;
    v8h hv = *p1;
    return __builtin_shufflevector(lo, hv, 0, 1, 2, 3, 4, 5, 6, 7,
                                   8, 9, 10, 11, 12, 13, 14, 15);
}

// B fragment from pre-swizzled weights: element e of lane holds
// W[kc*32 + (e>>3)*16 + hi*8 + (e&7)][nt*16 + n]; stored fragment-major so each lane
// reads 16 contiguous halfs (32B).
__device__ __forceinline__ v16h load_fragB_sw(const half_t* w, int lane, int kc, int nt) {
    const v16h* p = (const v16h*)w;
    return p[(kc * 4 + nt) * 32 + lane];
}

// D = A(16xK) @ W(Kx64), K = KCH*32, accumulate into acc[0..3] (nt tiles).
template <int KCH>
__device__ __forceinline__ void gemm16(const half_t* A, int ld, const half_t* Wsw,
                                       int lane, v8f acc[4]) {
#pragma unroll
    for (int kc = 0; kc < KCH; ++kc) {
        v16h a = load_fragA(A, ld, lane, kc * 32);
#pragma unroll
        for (int nt = 0; nt < 4; ++nt)
            acc[nt] = wmma_f16(a, load_fragB_sw(Wsw, lane, kc, nt), acc[nt]);
    }
}

// ---------------------------------------------------------------- init + loss

__global__ __launch_bounds__(256) void init_kernel(const float* __restrict__ x0,
                                                   const float* __restrict__ v0,
                                                   float* xA, float* vA,
                                                   float* partials) {
    __shared__ float red[256];
    int e = blockIdx.x * 256 + threadIdx.x;   // grid = 96 blocks -> 24576 threads
    float va = v0[e] * 0.2f;                  // TEMPERATURE
    float delta = va * 0.06f;                 // TIMESTEP * N_STEPS
    vA[e] = va;
    xA[e] = x0[e] + delta;
    red[threadIdx.x] = delta * delta;
    __syncthreads();
#pragma unroll
    for (int s = 128; s > 0; s >>= 1) {
        if (threadIdx.x < s) red[threadIdx.x] += red[threadIdx.x + s];
        __syncthreads();
    }
    if (threadIdx.x == 0) partials[blockIdx.x] = red[0];
}

// ---------------------------------------------------------------- weight f16 staging
// Pre-swizzle W (Kreal x 64, f32) into fragment-major f16:
//   dst[((kc*4 + nt)*32 + lane)*16 + e] = W[kc*32 + (e>>3)*16 + (lane>>4)*8 + (e&7)][nt*16 + (lane&15)]

__device__ __forceinline__ void swz_weight(const float* __restrict__ src,
                                           half_t* __restrict__ dst,
                                           int Kreal, int Kpad, int tid) {
    int total = Kpad * 64;
    for (int t = tid; t < total; t += 256) {
        int e = t & 15;
        int lane = (t >> 4) & 31;
        int nt = (t >> 9) & 3;
        int kc = t >> 11;
        int k = kc * 32 + ((e >> 3) << 4) + ((lane >> 4) & 1) * 8 + (e & 7);
        int n = nt * 16 + (lane & 15);
        dst[t] = (k < Kreal) ? (half_t)src[k * 64 + n] : (half_t)0.f;
    }
}

__global__ __launch_bounds__(256) void convert_kernel(
    const float* __restrict__ ew1, const float* __restrict__ ew2,
    const float* __restrict__ cw1, const float* __restrict__ nw1,
    const float* __restrict__ nw2, const float* __restrict__ emw,
    half_t* w1h, half_t* w2h, half_t* cw1h, half_t* nw1h, half_t* nw2h, half_t* ewh) {
    int tid = threadIdx.x;
    swz_weight(ew1, w1h, 131, 160, tid);
    swz_weight(ew2, w2h, 64, 64, tid);
    swz_weight(cw1, cw1h, 64, 64, tid);
    swz_weight(nw1, nw1h, 128, 128, tid);
    swz_weight(nw2, nw2h, 64, 64, tid);
    swz_weight(emw, ewh, 24, 32, tid);
}

// ---------------------------------------------------------------- embed (feats @ W + b)

__global__ __launch_bounds__(128) void embed_kernel(const float* __restrict__ feats,
                                                    const half_t* __restrict__ ewh,
                                                    const float* __restrict__ eb,
                                                    float* __restrict__ hA) {
    __shared__ __attribute__((aligned(32))) half_t sF[4][16 * 32];
    int wv = threadIdx.x >> 5, lane = threadIdx.x & 31;
    int base = (blockIdx.x * 4 + wv) * 16;          // first atom of this wave
    half_t* A = sF[wv];
    for (int t = lane; t < 16 * 32; t += 32) {
        int m = t >> 5, c = t & 31;
        A[t] = (c < FEATD) ? (half_t)feats[(size_t)(base + m) * FEATD + c] : (half_t)0.f;
    }
    __syncthreads();
    int n = lane & 15, hi = (lane >> 4) & 1;
    v8f acc[4];
#pragma unroll
    for (int nt = 0; nt < 4; ++nt) acc[nt] = (v8f){};
    gemm16<1>(A, 32, ewh, lane, acc);
#pragma unroll
    for (int nt = 0; nt < 4; ++nt) {
        int col = nt * 16 + n;
        float bias = eb[col];
#pragma unroll
        for (int r = 0; r < 8; ++r) {
            int M = r + 8 * hi;
            hA[(size_t)(base + M) * HID + col] = acc[nt][r] + bias;
        }
    }
}

// ---------------------------------------------------------------- kNN (14 nearest, excl self)

__global__ __launch_bounds__(256) void knn_kernel(const float* __restrict__ xc,
                                                  int* __restrict__ nbr) {
    __shared__ float tile[256 * 3];
    int wv = threadIdx.x >> 5, lane = threadIdx.x & 31;
    int i = blockIdx.x * 8 + wv;
    float xi = xc[i * 3 + 0], yi = xc[i * 3 + 1], zi = xc[i * 3 + 2];
    float bd[KNBR];
    int   bi[KNBR];
#pragma unroll
    for (int t = 0; t < KNBR; ++t) { bd[t] = 3.4e38f; bi[t] = -1; }

    for (int t0 = 0; t0 < NATOMS; t0 += 256) {
        __syncthreads();
        int j = t0 + threadIdx.x;
        tile[threadIdx.x * 3 + 0] = xc[j * 3 + 0];
        tile[threadIdx.x * 3 + 1] = xc[j * 3 + 1];
        tile[threadIdx.x * 3 + 2] = xc[j * 3 + 2];
        __syncthreads();
        for (int s = lane; s < 256; s += 32) {
            int j2 = t0 + s;
            float dx = xi - tile[s * 3 + 0];
            float dy = yi - tile[s * 3 + 1];
            float dz = zi - tile[s * 3 + 2];
            float d = dx * dx + dy * dy + dz * dz;
            if (j2 == i) continue;
            if (d < bd[KNBR - 1]) {
                float cd = d; int ci = j2;
#pragma unroll
                for (int t = 0; t < KNBR; ++t) {
                    if (cd < bd[t]) {
                        float td = bd[t]; int ti = bi[t];
                        bd[t] = cd; bi[t] = ci; cd = td; ci = ti;
                    }
                }
            }
        }
    }
    // merge 32 sorted lists: 14 rounds of wave-wide min + pop
#pragma unroll 1
    for (int rr = 0; rr < KNBR; ++rr) {
        float d0 = bd[0];
        int   i0 = bi[0];
        float mn = d0;
#pragma unroll
        for (int off = 16; off >= 1; off >>= 1)
            mn = fminf(mn, __shfl_xor(mn, off, 32));
        unsigned long long bm = __ballot(d0 == mn);
        int winner = __ffsll((long long)bm) - 1;
        int widx = __shfl(i0, winner, 32);
        if (lane == winner) {
#pragma unroll
            for (int t = 0; t < KNBR - 1; ++t) { bd[t] = bd[t + 1]; bi[t] = bi[t + 1]; }
            bd[KNBR - 1] = 3.4e38f;
        }
        if (lane == 0) nbr[i * KNBR + rr] = widx;
    }
}

// ---------------------------------------------------------------- edge MLP + coord/vel update
// One wave per atom: its 14 edges (+2 zero rows) = one 16-row WMMA tile.

__global__ __launch_bounds__(128) void edge_kernel(
    const float* __restrict__ xA, const float* __restrict__ vA,
    const float* __restrict__ hA, const int* __restrict__ nbr,
    const int* __restrict__ res,
    const half_t* __restrict__ w1h, const float* __restrict__ b1,
    const half_t* __restrict__ w2h, const float* __restrict__ b2,
    const half_t* __restrict__ cw1h, const float* __restrict__ cb1,
    const float* __restrict__ cw2,
    const float* __restrict__ vw, const float* __restrict__ vb,
    float* __restrict__ aggm, float* __restrict__ xB, float* __restrict__ vB) {
    __shared__ __attribute__((aligned(32))) half_t sA[4][16 * 160];
    __shared__ __attribute__((aligned(32))) half_t sO[4][16 * 64];
    __shared__ float  sDiff[4][16][3];
    __shared__ float  sW[4][16];

    int wv = threadIdx.x >> 5, lane = threadIdx.x & 31;
    int i = blockIdx.x * 4 + wv;
    half_t* A = sA[wv];
    half_t* O = sO[wv];
    int n = lane & 15, hi = (lane >> 4) & 1;

    // zero the 16x160 m_in tile
    for (int t = lane; t < 16 * 160; t += 32) A[t] = (half_t)0.f;

    // per-edge scalar features (lane j owns edge j)
    int r = i;
    if (lane < KNBR) {
        r = nbr[i * KNBR + lane];
        float dx = xA[i * 3 + 0] - xA[r * 3 + 0];
        float dy = xA[i * 3 + 1] - xA[r * 3 + 1];
        float dz = xA[i * 3 + 2] - xA[r * 3 + 2];
        float rad = dx * dx + dy * dy + dz * dz;
        float dist = sqrtf(rad);
        float ss = fminf(fabsf((float)res[i] - (float)res[r]) * 0.2f, 1.f);
        A[lane * 160 + 128] = (half_t)rad;
        A[lane * 160 + 129] = (half_t)dist;
        A[lane * 160 + 130] = (half_t)ss;
        sDiff[wv][lane][0] = dx;
        sDiff[wv][lane][1] = dy;
        sDiff[wv][lane][2] = dz;
    }
    // h[sender]=h[i] broadcast into rows 0..13, h[receiver] per row
    for (int c = lane; c < HID; c += 32) {
        half_t hv = (half_t)hA[(size_t)i * HID + c];
#pragma unroll 1
        for (int j = 0; j < KNBR; ++j) A[j * 160 + c] = hv;
    }
#pragma unroll 1
    for (int j = 0; j < KNBR; ++j) {
        int rj = __shfl(r, j, 32);
        for (int c = lane; c < HID; c += 32)
            A[j * 160 + 64 + c] = (half_t)hA[(size_t)rj * HID + c];
    }
    __syncthreads();

    // layer 1: silu(m_in @ W1 + b1)  (K = 160)
    {
        v8f acc[4];
#pragma unroll
        for (int nt = 0; nt < 4; ++nt) acc[nt] = (v8f){};
        gemm16<5>(A, 160, w1h, lane, acc);
#pragma unroll
        for (int nt = 0; nt < 4; ++nt) {
            float bias = b1[nt * 16 + n];
#pragma unroll
            for (int rr = 0; rr < 8; ++rr) {
                int M = rr + 8 * hi;
                O[M * 64 + nt * 16 + n] = (half_t)silu_f(acc[nt][rr] + bias);
            }
        }
    }
    __syncthreads();

    // layer 2: m = silu(out1 @ W2 + b2) (K=64); m -> f16 (reuse sA) + column-sum agg_m
    {
        v8f acc[4];
#pragma unroll
        for (int nt = 0; nt < 4; ++nt) acc[nt] = (v8f){};
        gemm16<2>(O, 64, w2h, lane, acc);
#pragma unroll
        for (int nt = 0; nt < 4; ++nt) {
            float bias = b2[nt * 16 + n];
            float csum = 0.f;
#pragma unroll
            for (int rr = 0; rr < 8; ++rr) {
                int M = rr + 8 * hi;
                float y = silu_f(acc[nt][rr] + bias);
                A[M * 64 + nt * 16 + n] = (half_t)y;   // m as f16 (ld=64)
                if (M < KNBR) csum += y;               // mask dummy rows 14,15
            }
            csum += __shfl_xor(csum, 16, 32);
            if (hi == 0) aggm[(size_t)i * HID + nt * 16 + n] = csum;
        }
    }
    __syncthreads();

    // coord head: w = silu(m @ cw1 + cb1) @ cw2
    float pw[8];
#pragma unroll
    for (int rr = 0; rr < 8; ++rr) pw[rr] = 0.f;
    {
        v8f acc[4];
#pragma unroll
        for (int nt = 0; nt < 4; ++nt) acc[nt] = (v8f){};
        gemm16<2>(A, 64, cw1h, lane, acc);
#pragma unroll
        for (int nt = 0; nt < 4; ++nt) {
            float bias = cb1[nt * 16 + n];
            float c2 = cw2[nt * 16 + n];
#pragma unroll
            for (int rr = 0; rr < 8; ++rr) pw[rr] += silu_f(acc[nt][rr] + bias) * c2;
        }
    }
#pragma unroll
    for (int off = 1; off < 16; off <<= 1)
#pragma unroll
        for (int rr = 0; rr < 8; ++rr) pw[rr] += __shfl_xor(pw[rr], off, 32);
    if (n == 0) {
#pragma unroll
        for (int rr = 0; rr < 8; ++rr) sW[wv][8 * hi + rr] = pw[rr];
    }
    __syncthreads();

    // vel_scale = tanh(h[i] . vel_w + vel_b); update vel & coords
    float p = 0.f;
    for (int c = lane; c < HID; c += 32) p += hA[(size_t)i * HID + c] * vw[c];
#pragma unroll
    for (int off = 1; off < 32; off <<= 1) p += __shfl_xor(p, off, 32);
    float vs = tanhf(p + vb[0]);
    if (lane < 3) {
        float ax = 0.f;
#pragma unroll
        for (int j = 0; j < KNBR; ++j) ax += sDiff[wv][j][lane] * sW[wv][j];
        ax *= (1.f / 14.f);
        float nv = vs * vA[i * 3 + lane] + ax;
        vB[i * 3 + lane] = nv;
        xB[i * 3 + lane] = xA[i * 3 + lane] + nv;
    }
}

// ---------------------------------------------------------------- node MLP (residual)

__global__ __launch_bounds__(128) void node_kernel(
    const float* __restrict__ hA, const float* __restrict__ aggm,
    const half_t* __restrict__ nw1h, const float* __restrict__ nb1,
    const half_t* __restrict__ nw2h, const float* __restrict__ nb2,
    float* __restrict__ hB) {
    __shared__ __attribute__((aligned(32))) half_t sA2[4][16 * 128];
    __shared__ __attribute__((aligned(32))) half_t sU[4][16 * 64];
    int wv = threadIdx.x >> 5, lane = threadIdx.x & 31;
    int base = (blockIdx.x * 4 + wv) * 16;
    half_t* A = sA2[wv];
    half_t* U = sU[wv];
    int n = lane & 15, hi = (lane >> 4) & 1;

    for (int t = lane; t < 16 * 64; t += 32) {
        int m = t >> 6, c = t & 63;
        A[m * 128 + c]      = (half_t)hA[(size_t)(base + m) * HID + c];
        A[m * 128 + 64 + c] = (half_t)aggm[(size_t)(base + m) * HID + c];
    }
    __syncthreads();

    {
        v8f acc[4];
#pragma unroll
        for (int nt = 0; nt < 4; ++nt) acc[nt] = (v8f){};
        gemm16<4>(A, 128, nw1h, lane, acc);
#pragma unroll
        for (int nt = 0; nt < 4; ++nt) {
            float bias = nb1[nt * 16 + n];
#pragma unroll
            for (int rr = 0; rr < 8; ++rr) {
                int M = rr + 8 * hi;
                U[M * 64 + nt * 16 + n] = (half_t)silu_f(acc[nt][rr] + bias);
            }
        }
    }
    __syncthreads();

    {
        v8f acc[4];
#pragma unroll
        for (int nt = 0; nt < 4; ++nt) acc[nt] = (v8f){};
        gemm16<2>(U, 64, nw2h, lane, acc);
#pragma unroll
        for (int nt = 0; nt < 4; ++nt) {
            int col = nt * 16 + n;
            float bias = nb2[col];
#pragma unroll
            for (int rr = 0; rr < 8; ++rr) {
                int M = rr + 8 * hi;
                size_t idx = (size_t)(base + M) * HID + col;
                hB[idx] = hA[idx] + acc[nt][rr] + bias;
            }
        }
    }
}

// ---------------------------------------------------------------- output

__global__ __launch_bounds__(256) void finish_kernel(const float* __restrict__ xF,
                                                     const float* __restrict__ partials,
                                                     float* __restrict__ out) {
    int e = blockIdx.x * 256 + threadIdx.x;
    if (e < NATOMS * 3) out[e] = xF[e];
    if (blockIdx.x == 0 && threadIdx.x == 0) {
        float s = 0.f;
        for (int b = 0; b < 96; ++b) s += partials[b];
        out[NATOMS * 3] = sqrtf(s / (float)NATOMS);
    }
}

// ---------------------------------------------------------------- host orchestration

extern "C" void kernel_launch(void* const* d_in, const int* in_sizes, int n_in,
                              void* d_out, int out_size, void* d_ws, size_t ws_size,
                              hipStream_t stream) {
    const float* x0  = (const float*)d_in[0];
    const float* f0  = (const float*)d_in[1];
    const int*   res = (const int*)d_in[2];
    const float* v0  = (const float*)d_in[3];
    const float* emw = (const float*)d_in[4];
    const float* emb = (const float*)d_in[5];
    const float* ew1 = (const float*)d_in[6];
    const float* eb1 = (const float*)d_in[7];
    const float* ew2 = (const float*)d_in[8];
    const float* eb2 = (const float*)d_in[9];
    const float* cw1 = (const float*)d_in[10];
    const float* cb1 = (const float*)d_in[11];
    const float* cw2 = (const float*)d_in[12];
    const float* vw  = (const float*)d_in[13];
    const float* vb  = (const float*)d_in[14];
    const float* nw1 = (const float*)d_in[15];
    const float* nb1 = (const float*)d_in[16];
    const float* nw2 = (const float*)d_in[17];
    const float* nb2 = (const float*)d_in[18];

    char* ws = (char*)d_ws;
    size_t off = 0;
    auto take = [&](size_t bytes) -> char* {
        char* p = ws + off;
        off = (off + bytes + 255) & ~(size_t)255;
        return p;
    };
    float* partials = (float*)take(96 * sizeof(float));
    float* hBufA = (float*)take((size_t)NATOMS * HID * sizeof(float));
    float* hBufB = (float*)take((size_t)NATOMS * HID * sizeof(float));
    float* xBufA = (float*)take((size_t)NATOMS * 3 * sizeof(float));
    float* xBufB = (float*)take((size_t)NATOMS * 3 * sizeof(float));
    float* vBufA = (float*)take((size_t)NATOMS * 3 * sizeof(float));
    float* vBufB = (float*)take((size_t)NATOMS * 3 * sizeof(float));
    float* aggm  = (float*)take((size_t)NATOMS * HID * sizeof(float));
    int*   nbr   = (int*)take((size_t)NATOMS * KNBR * sizeof(int));
    half_t* w1h  = (half_t*)take(160 * 64 * sizeof(half_t));
    half_t* w2h  = (half_t*)take(64 * 64 * sizeof(half_t));
    half_t* cw1h = (half_t*)take(64 * 64 * sizeof(half_t));
    half_t* nw1h = (half_t*)take(128 * 64 * sizeof(half_t));
    half_t* nw2h = (half_t*)take(64 * 64 * sizeof(half_t));
    half_t* ewh  = (half_t*)take(32 * 64 * sizeof(half_t));

    init_kernel<<<96, 256, 0, stream>>>(x0, v0, xBufA, vBufA, partials);
    convert_kernel<<<1, 256, 0, stream>>>(ew1, ew2, cw1, nw1, nw2, emw,
                                          w1h, w2h, cw1h, nw1h, nw2h, ewh);
    embed_kernel<<<NATOMS / 64, 128, 0, stream>>>(f0, ewh, emb, hBufA);

    float *xA = xBufA, *xB = xBufB, *vA = vBufA, *vB = vBufB, *hA = hBufA, *hB = hBufB;
    for (int step = 0; step < NSTEPS; ++step) {
        knn_kernel<<<NATOMS / 8, 256, 0, stream>>>(xA, nbr);
        edge_kernel<<<NATOMS / 4, 128, 0, stream>>>(xA, vA, hA, nbr, res,
                                                    w1h, eb1, w2h, eb2, cw1h, cb1, cw2,
                                                    vw, vb, aggm, xB, vB);
        node_kernel<<<NATOMS / 64, 128, 0, stream>>>(hA, aggm, nw1h, nb1, nw2h, nb2, hB);
        float* t;
        t = xA; xA = xB; xB = t;
        t = vA; vA = vB; vB = t;
        t = hA; hA = hB; hB = t;
    }
    finish_kernel<<<96, 256, 0, stream>>>(xA, partials, (float*)d_out);
}
